// GNN_node_68573447848168
// MI455X (gfx1250) — matrix operand order repaired and checked by reading, not compiled, not run on hardware
//
#include <hip/hip_runtime.h>

#define LNUM 5
#define NNODE 50000
#define NEDGE 400000
#define DDIM 128
#define HDIM 256
#define BN_EPS 1e-5f

typedef __bf16 v16bf __attribute__((ext_vector_type(16)));
typedef float  v8f   __attribute__((ext_vector_type(8)));
typedef unsigned int u32x4 __attribute__((ext_vector_type(4)));

union FragU { v16bf v; u32x4 q[2]; };

static __device__ __forceinline__ unsigned short f2bf(float f) {
  unsigned int u = __float_as_uint(f);
  unsigned int r = (u + 0x7FFFu + ((u >> 16) & 1u)) >> 16;
  return (unsigned short)r;
}

// -------- 1. fused bond-embedding + gather + relu + scatter-add ------------
// one wave (32 lanes) per edge; each lane owns 4 consecutive floats of D=128
__global__ void edge_scatter(const float* __restrict__ h,
                             const int* __restrict__ ei,      // [2,E]
                             const int* __restrict__ ea,      // [E,3]
                             const float* __restrict__ bond,  // [13,D] (this layer)
                             float* __restrict__ agg) {
  int gid  = blockIdx.x * blockDim.x + threadIdx.x;
  int e    = gid >> 5;
  int lane = gid & 31;
  if (e >= NEDGE) return;
  int src = ei[e];
  int dst = ei[NEDGE + e];
  int i0 = ea[e * 3 + 0] + 0;
  int i1 = ea[e * 3 + 1] + 5;
  int i2 = ea[e * 3 + 2] + 11;
  int d = lane * 4;
  float4 hv = *(const float4*)(h + (size_t)src * DDIM + d);
  float4 e0 = *(const float4*)(bond + (size_t)i0 * DDIM + d);
  float4 e1 = *(const float4*)(bond + (size_t)i1 * DDIM + d);
  float4 e2 = *(const float4*)(bond + (size_t)i2 * DDIM + d);
  float m0 = fmaxf(hv.x + e0.x + e1.x + e2.x, 0.f);
  float m1 = fmaxf(hv.y + e0.y + e1.y + e2.y, 0.f);
  float m2 = fmaxf(hv.z + e0.z + e1.z + e2.z, 0.f);
  float m3 = fmaxf(hv.w + e0.w + e1.w + e2.w, 0.f);
  float* a = agg + (size_t)dst * DDIM + d;
  atomicAdd(a + 0, m0);
  atomicAdd(a + 1, m1);
  atomicAdd(a + 2, m2);
  atomicAdd(a + 3, m3);
}

// -------- 2. z = (1+eps)*h + agg  ->  bf16 ---------------------------------
__global__ void prep_z(const float* __restrict__ h, const float* __restrict__ agg,
                       const float* __restrict__ eps_arr, int layer,
                       unsigned short* __restrict__ z, int total) {
  float e = 1.0f + eps_arr[layer];
  for (int i = blockIdx.x * blockDim.x + threadIdx.x; i < total;
       i += gridDim.x * blockDim.x) {
    z[i] = f2bf(e * h[i] + agg[i]);
  }
}

// -------- 3. weight transpose + bf16 convert: W[K,Nout] -> Wt[Nout,K] ------
__global__ void convw(const float* __restrict__ w, unsigned short* __restrict__ wt,
                      int K, int Nout) {
  int i = blockIdx.x * blockDim.x + threadIdx.x;
  if (i >= K * Nout) return;
  int k = i / Nout;
  int n = i - k * Nout;
  wt[(size_t)n * K + k] = f2bf(w[i]);
}

// -------- 4. WMMA GEMM: C[M,Nout] = A[M,K](bf16) * Bt[Nout,K](bf16) + bias -
// one wave per 16x16 output tile; K stepped by 32 (v_wmma_f32_16x16x32_bf16)
__global__ void wmma_gemm(const unsigned short* __restrict__ A,
                          const unsigned short* __restrict__ Bt,
                          const float* __restrict__ bias,
                          float* __restrict__ C,
                          int M, int K, int Nout, int tiles) {
  int wave = (blockIdx.x * blockDim.x + threadIdx.x) >> 5;
  if (wave >= tiles) return;                 // wave-uniform
  int lane   = threadIdx.x & 31;
  int tilesN = Nout >> 4;
  int tm = wave / tilesN;
  int tn = wave - tm * tilesN;

  int hi  = lane >> 4;                       // lane half selects K sub-runs
  int l15 = lane & 15;
  // A frag: lanes 0-15 -> K {0..7,16..23}; lanes 16-31 -> K {8..15,24..31}
  const unsigned short* aRow = A + (size_t)(tm * 16 + l15) * K + (hi ? 8 : 0);
  // B frag: lanes 0-15 -> K {0..7,16..23}; lanes 16-31 -> K {8..15,24..31} (col = l15)
  const unsigned short* bRow = Bt + (size_t)(tn * 16 + l15) * K + (hi ? 8 : 0);

  v8f acc = {};
  for (int k0 = 0; k0 < K; k0 += 32) {
    FragU a, b;
    a.q[0] = *(const u32x4*)(aRow + k0);
    a.q[1] = *(const u32x4*)(aRow + k0 + 16);
    b.q[0] = *(const u32x4*)(bRow + k0);
    b.q[1] = *(const u32x4*)(bRow + k0 + 16);
    acc = __builtin_amdgcn_wmma_f32_16x16x32_bf16(
        false, a.v, false, b.v, (short)0, acc, false, false);
  }
  int col = tn * 16 + l15;
  float bv = bias[col];
  for (int r = 0; r < 8; ++r) {
    int row = tm * 16 + (hi ? 8 + r : r);
    C[(size_t)row * Nout + col] = acc[r] + bv;
  }
}

// -------- 5. BN column stats: sum / sumsq per column via atomics -----------
__global__ void bn_stats(const float* __restrict__ y, int n, int c,
                         float* __restrict__ sum, float* __restrict__ sq) {
  int col = threadIdx.x;                     // blockDim.x == c
  float s = 0.f, s2 = 0.f;
  for (int r = blockIdx.x; r < n; r += gridDim.x) {
    float v = y[(size_t)r * c + col];
    s += v;
    s2 += v * v;
  }
  atomicAdd(&sum[col], s);
  atomicAdd(&sq[col], s2);
}

// -------- 6. BN + relu -> bf16 (inner MLP activation) ----------------------
__global__ void bn_relu_bf16(const float* __restrict__ y,
                             const float* __restrict__ sum, const float* __restrict__ sq,
                             const float* __restrict__ g, const float* __restrict__ b,
                             unsigned short* __restrict__ out, int n, int c) {
  float inv_n = 1.0f / (float)n;
  for (int i = blockIdx.x * blockDim.x + threadIdx.x; i < n * c;
       i += gridDim.x * blockDim.x) {
    int col = i % c;
    float mean = sum[col] * inv_n;
    float var  = sq[col] * inv_n - mean * mean;
    float sc   = g[col] * rsqrtf(var + BN_EPS);
    float v    = sc * (y[i] - mean) + b[col];
    out[i] = f2bf(fmaxf(v, 0.f));
  }
}

// -------- 7. outer BN (+optional relu) -> f32 h_out ------------------------
__global__ void bn_final(const float* __restrict__ y,
                         const float* __restrict__ sum, const float* __restrict__ sq,
                         const float* __restrict__ g, const float* __restrict__ b,
                         float* __restrict__ out, int n, int c, int do_relu) {
  float inv_n = 1.0f / (float)n;
  for (int i = blockIdx.x * blockDim.x + threadIdx.x; i < n * c;
       i += gridDim.x * blockDim.x) {
    int col = i % c;
    float mean = sum[col] * inv_n;
    float var  = sq[col] * inv_n - mean * mean;
    float sc   = g[col] * rsqrtf(var + BN_EPS);
    float v    = sc * (y[i] - mean) + b[col];
    out[i] = do_relu ? fmaxf(v, 0.f) : v;
  }
}

extern "C" void kernel_launch(void* const* d_in, const int* in_sizes, int n_in,
                              void* d_out, int out_size, void* d_ws, size_t ws_size,
                              hipStream_t stream) {
  const float* x    = (const float*)d_in[0];
  const int*   ei   = (const int*)d_in[1];
  const int*   ea   = (const int*)d_in[2];
  const float* W1   = (const float*)d_in[3];
  const float* b1   = (const float*)d_in[4];
  const float* g1   = (const float*)d_in[5];
  const float* bb1  = (const float*)d_in[6];
  const float* W2   = (const float*)d_in[7];
  const float* b2   = (const float*)d_in[8];
  const float* eps  = (const float*)d_in[9];
  const float* bond = (const float*)d_in[10];
  const float* g2   = (const float*)d_in[11];
  const float* bb2  = (const float*)d_in[12];

  char* ws = (char*)d_ws;
  float* h_buf = (float*)ws;                 ws += (size_t)NNODE * DDIM * 4;   // 25.6MB
  float* agg   = (float*)ws;                 ws += (size_t)NNODE * DDIM * 4;   // 25.6MB (reused as y2)
  unsigned short* z_bf = (unsigned short*)ws; ws += (size_t)NNODE * DDIM * 2;  // 12.8MB
  float* y1    = (float*)ws;                 ws += (size_t)NNODE * HDIM * 4;   // 51.2MB
  unsigned short* a1_bf = (unsigned short*)ws; ws += (size_t)NNODE * HDIM * 2; // 25.6MB
  unsigned short* wt1 = (unsigned short*)ws; ws += (size_t)DDIM * HDIM * 2;    // 64KB
  unsigned short* wt2 = (unsigned short*)ws; ws += (size_t)HDIM * DDIM * 2;    // 64KB
  float* stat_sum = (float*)ws;              ws += 256 * 4;
  float* stat_sq  = (float*)ws;              ws += 256 * 4;
  float* y2 = agg;   // agg is dead once z is built; reuse for GEMM2 output

  const int ND = NNODE * DDIM;
  const int NH = NNODE * HDIM;

  for (int l = 0; l < LNUM; ++l) {
    const float* h_in = (l == 0) ? x : h_buf;
    float* h_out = (l == LNUM - 1) ? (float*)d_out : h_buf;

    // aggregate messages
    hipMemsetAsync(agg, 0, (size_t)ND * sizeof(float), stream);
    edge_scatter<<<(NEDGE * 32) / 256, 256, 0, stream>>>(
        h_in, ei, ea, bond + (size_t)l * 13 * DDIM, agg);

    // z = (1+eps)*h + agg -> bf16
    prep_z<<<2048, 256, 0, stream>>>(h_in, agg, eps, l, z_bf, ND);

    // weights -> transposed bf16
    convw<<<(DDIM * HDIM + 255) / 256, 256, 0, stream>>>(
        W1 + (size_t)l * DDIM * HDIM, wt1, DDIM, HDIM);
    convw<<<(HDIM * DDIM + 255) / 256, 256, 0, stream>>>(
        W2 + (size_t)l * HDIM * DDIM, wt2, HDIM, DDIM);

    // GEMM1: y1[N,H] = z[N,D] * W1 + b1
    {
      int tiles = (NNODE / 16) * (HDIM / 16);
      int blocks = (tiles * 32 + 255) / 256;
      wmma_gemm<<<blocks, 256, 0, stream>>>(z_bf, wt1, b1 + (size_t)l * HDIM,
                                            y1, NNODE, DDIM, HDIM, tiles);
    }
    // BN(inner) + relu -> bf16
    hipMemsetAsync(stat_sum, 0, 2 * 256 * sizeof(float), stream);
    bn_stats<<<256, HDIM, 0, stream>>>(y1, NNODE, HDIM, stat_sum, stat_sq);
    bn_relu_bf16<<<4096, 256, 0, stream>>>(y1, stat_sum, stat_sq,
                                           g1 + (size_t)l * HDIM,
                                           bb1 + (size_t)l * HDIM, a1_bf,
                                           NNODE, HDIM);

    // GEMM2: y2[N,D] = a1[N,H] * W2 + b2
    {
      int tiles = (NNODE / 16) * (DDIM / 16);
      int blocks = (tiles * 32 + 255) / 256;
      wmma_gemm<<<blocks, 256, 0, stream>>>(a1_bf, wt2, b2 + (size_t)l * DDIM,
                                            y2, NNODE, HDIM, DDIM, tiles);
    }
    // outer BN (+relu except last layer)
    hipMemsetAsync(stat_sum, 0, 2 * 256 * sizeof(float), stream);
    bn_stats<<<256, DDIM, 0, stream>>>(y2, NNODE, DDIM, stat_sum, stat_sq);
    bn_final<<<4096, 256, 0, stream>>>(y2, stat_sum, stat_sq,
                                       g2 + (size_t)l * DDIM,
                                       bb2 + (size_t)l * DDIM, h_out,
                                       NNODE, DDIM, (l < LNUM - 1) ? 1 : 0);
  }
  (void)in_sizes; (void)n_in; (void)out_size; (void)ws_size;
}
